// MultiHeadAttention_77713138254073
// MI455X (gfx1250) — compile-verified, matrix operand order checked by reading, not run on hardware
//
#include <hip/hip_runtime.h>

typedef __attribute__((ext_vector_type(16))) __bf16 v16bf;
typedef __attribute__((ext_vector_type(8)))  __bf16 v8bf;
typedef __attribute__((ext_vector_type(8)))  float  v8f;

#define DIM    768
#define SEQ    2048
#define NHEAD  12
#define HDIM   64
#define NBATCH 2
#define NTOK   4096   /* NBATCH*SEQ */

__device__ __forceinline__ unsigned short f2bf(float f) {
  unsigned int u = __builtin_bit_cast(unsigned int, f);
  unsigned int r = u + 0x7FFFu + ((u >> 16) & 1u);   // round-to-nearest-even
  return (unsigned short)(r >> 16);
}

__device__ __forceinline__ v8f zero8() {
  v8f z;
#pragma unroll
  for (int i = 0; i < 8; ++i) z[i] = 0.0f;
  return z;
}

__device__ __forceinline__ v8f wmma_bf16(v16bf a, v16bf b, v8f c) {
  return __builtin_amdgcn_wmma_f32_16x16x32_bf16(false, a, false, b, (short)0, c, false, false);
}

// CDNA5 16-bit A/B operand gather (16x32 M-major or 32x16 K-major, same pattern):
// lane<16 : row = base+lane,    K in {k0..k0+7, k0+16..k0+23}
// lane>=16: row = base+lane-16, K in {k0+8..k0+15, k0+24..k0+31}
__device__ __forceinline__ v16bf load_op16(const unsigned short* base, int stride,
                                           int row, int k0) {
  const int lane = threadIdx.x & 31;
  const int ph   = lane >> 4;
  const int li   = lane & 15;
  const unsigned short* p = base + (size_t)(row + li) * stride + k0 + ph * 8;
  v8bf lo = *reinterpret_cast<const v8bf*>(p);
  v8bf hi = *reinterpret_cast<const v8bf*>(p + 16);
  v16bf r;
#pragma unroll
  for (int i = 0; i < 8; ++i) { r[i] = lo[i]; r[i + 8] = hi[i]; }
  return r;
}

__global__ __launch_bounds__(256) void f2bf_kernel(const float* __restrict__ in,
                                                   unsigned short* __restrict__ out,
                                                   int n) {
  int i = blockIdx.x * 256 + threadIdx.x;
  if (i < n) out[i] = f2bf(in[i]);
}

// Writes one 16x16 C tile. mode 0: bf16 [b][h][s][d]; mode 1: bf16 [b][h][d][s]; mode 2: f32 row-major.
__device__ __forceinline__ void store_tile(const v8f& c, int rowBase, int colBase,
                                           const float* __restrict__ bias,
                                           float* __restrict__ outF,
                                           unsigned short* __restrict__ outH, int mode) {
  const int lane = threadIdx.x & 31;
  const int ph = lane >> 4, li = lane & 15;
#pragma unroll
  for (int j = 0; j < 8; ++j) {
    const int r  = rowBase + j + ph * 8;
    const int cg = colBase + li;
    const float v = c[j] + bias[cg];
    if (mode == 2) {
      outF[(size_t)r * DIM + cg] = v;
    } else {
      const int b = r >> 11, s = r & (SEQ - 1);
      const int h = cg >> 6, d = cg & (HDIM - 1);
      const size_t addr = (mode == 0)
          ? ((size_t)(b * NHEAD + h) * SEQ + s) * HDIM + d
          : ((size_t)(b * NHEAD + h) * HDIM + d) * SEQ + s;
      outH[addr] = f2bf(v);
    }
  }
}

// out(r,c) = sum_k X[r,k]*W[c,k] + bias[c].  M=4096, N=K=768.
// 64x256 block tile (8 waves as 2x4), 32x64 per wave: 12 b128 loads per 8 WMMAs.
__global__ __launch_bounds__(256) void gemm_bf16_kernel(
    const unsigned short* __restrict__ X, const unsigned short* __restrict__ W,
    const float* __restrict__ bias, float* __restrict__ outF,
    unsigned short* __restrict__ outH, int mode) {
  const int w  = threadIdx.x >> 5;
  const int wm = w >> 2;          // 0..1
  const int wn = w & 3;           // 0..3
  const int rowBase = blockIdx.x * 64  + wm * 32;
  const int colBase = blockIdx.y * 256 + wn * 64;
  v8f c0[4], c1[4];
#pragma unroll
  for (int t = 0; t < 4; ++t) { c0[t] = zero8(); c1[t] = zero8(); }

  for (int k = 0; k < DIM; k += 32) {
    if (k + 64 < DIM) {  // prefetch two k-steps ahead -> global_prefetch_b8
      const int lane = threadIdx.x & 31;
      __builtin_prefetch(X + (size_t)(rowBase + (lane & 15)) * DIM + k + 64, 0, 1);
      __builtin_prefetch(W + (size_t)(colBase + (lane & 15)) * DIM + k + 64, 0, 1);
    }
    v16bf a0 = load_op16(X, DIM, rowBase,      k);
    v16bf a1 = load_op16(X, DIM, rowBase + 16, k);
    v16bf b0 = load_op16(W, DIM, colBase,      k);
    v16bf b1 = load_op16(W, DIM, colBase + 16, k);
    v16bf b2 = load_op16(W, DIM, colBase + 32, k);
    v16bf b3 = load_op16(W, DIM, colBase + 48, k);
    c0[0] = wmma_bf16(a0, b0, c0[0]);
    c1[0] = wmma_bf16(a1, b0, c1[0]);
    c0[1] = wmma_bf16(a0, b1, c0[1]);
    c1[1] = wmma_bf16(a1, b1, c1[1]);
    c0[2] = wmma_bf16(a0, b2, c0[2]);
    c1[2] = wmma_bf16(a1, b2, c1[2]);
    c0[3] = wmma_bf16(a0, b3, c0[3]);
    c1[3] = wmma_bf16(a1, b3, c1[3]);
  }
#pragma unroll
  for (int t = 0; t < 4; ++t) {
    store_tile(c0[t], rowBase,      colBase + 16 * t, bias, outF, outH, mode);
    store_tile(c1[t], rowBase + 16, colBase + 16 * t, bias, outF, outH, mode);
  }
}

// Flash-attention: one wave per 16 query rows, streaming K/V in 32-token chunks.
// Qp/Kp: bf16 [b][h][s][hd]; Vt: bf16 [b][h][hd][s]; concat: f32 scatter per reference reshape.
__global__ __launch_bounds__(128) void attn_kernel(
    const unsigned short* __restrict__ Qp, const unsigned short* __restrict__ Kp,
    const unsigned short* __restrict__ Vt, const unsigned char* __restrict__ mask,
    float* __restrict__ concat) {
  __shared__ unsigned short smP[4][16 * 32];
  const int wave = threadIdx.x >> 5;
  const int lane = threadIdx.x & 31;
  const int ph = lane >> 4, li = lane & 15;
  const int qb = blockIdx.x & 31;          // SEQ/64 = 32 q-blocks
  const int hb = blockIdx.x >> 5;
  const int h = hb / NBATCH, b = hb % NBATCH;
  const int q0 = qb * 64 + wave * 16;
  const unsigned short* Qh = Qp + (size_t)(b * NHEAD + h) * SEQ * HDIM;
  const unsigned short* Kh = Kp + (size_t)(b * NHEAD + h) * SEQ * HDIM;
  const unsigned short* Vh = Vt + (size_t)(b * NHEAD + h) * HDIM * SEQ;
  const float inv_scale = 0.036084391824351613f;  // 1/sqrt(768), source scales by sqrt(D)

  v16bf aQ0 = load_op16(Qh, HDIM, q0, 0);
  v16bf aQ1 = load_op16(Qh, HDIM, q0, 32);

  float Mr[8], Lr[8];
  v8f o[4];
#pragma unroll
  for (int j = 0; j < 8; ++j) { Mr[j] = -__builtin_inff(); Lr[j] = 0.0f; }
#pragma unroll
  for (int si = 0; si < 4; ++si) o[si] = zero8();

  for (int kb = 0; kb < SEQ; kb += 32) {
    v16bf b00 = load_op16(Kh, HDIM, kb,      0);
    v16bf b01 = load_op16(Kh, HDIM, kb,     32);
    v16bf b10 = load_op16(Kh, HDIM, kb + 16, 0);
    v16bf b11 = load_op16(Kh, HDIM, kb + 16, 32);
    v8f s0 = zero8(), s1 = zero8();
    s0 = wmma_bf16(aQ0, b00, s0); s0 = wmma_bf16(aQ1, b01, s0);
    s1 = wmma_bf16(aQ0, b10, s1); s1 = wmma_bf16(aQ1, b11, s1);

    float tmax[8];
#pragma unroll
    for (int j = 0; j < 8; ++j) {
      float x0 = s0[j] * inv_scale, x1 = s1[j] * inv_scale;
      const int q = q0 + j + ph * 8;
      const size_t mrow = ((size_t)b * SEQ + q) * SEQ;
      if (mask[mrow + kb + li])      x0 = -__builtin_inff();
      if (mask[mrow + kb + 16 + li]) x1 = -__builtin_inff();
      s0[j] = x0; s1[j] = x1;
      tmax[j] = fmaxf(x0, x1);
    }
#pragma unroll
    for (int j = 0; j < 8; ++j)
      for (int off = 1; off < 16; off <<= 1)
        tmax[j] = fmaxf(tmax[j], __shfl_xor(tmax[j], off, 32));

    float alpha[8], rs[8];
#pragma unroll
    for (int j = 0; j < 8; ++j) {
      const float nm = fmaxf(Mr[j], tmax[j]);
      alpha[j] = (Mr[j] == -__builtin_inff()) ? 0.0f : __expf(Mr[j] - nm);
      const float p0 = (s0[j] == -__builtin_inff()) ? 0.0f : __expf(s0[j] - nm);
      const float p1 = (s1[j] == -__builtin_inff()) ? 0.0f : __expf(s1[j] - nm);
      Mr[j] = nm; s0[j] = p0; s1[j] = p1;
      rs[j] = p0 + p1;
    }
#pragma unroll
    for (int j = 0; j < 8; ++j)
      for (int off = 1; off < 16; off <<= 1)
        rs[j] += __shfl_xor(rs[j], off, 32);
#pragma unroll
    for (int j = 0; j < 8; ++j) Lr[j] = Lr[j] * alpha[j] + rs[j];
#pragma unroll
    for (int si = 0; si < 4; ++si)
#pragma unroll
      for (int j = 0; j < 8; ++j) o[si][j] *= alpha[j];

    // P (C-layout) -> LDS -> A-layout bf16
    unsigned short* sm = smP[wave];
#pragma unroll
    for (int j = 0; j < 8; ++j) {
      const int m = j + ph * 8;
      sm[m * 32 + li]      = f2bf(s0[j]);
      sm[m * 32 + 16 + li] = f2bf(s1[j]);
    }
    __syncthreads();
    const unsigned short* pp = sm + li * 32 + ph * 8;
    v8bf plo = *reinterpret_cast<const v8bf*>(pp);
    v8bf phi = *reinterpret_cast<const v8bf*>(pp + 16);
    v16bf aP;
#pragma unroll
    for (int i = 0; i < 8; ++i) { aP[i] = plo[i]; aP[i + 8] = phi[i]; }

#pragma unroll
    for (int si = 0; si < 4; ++si) {
      v16bf bv = load_op16(Vh, SEQ, si * 16, kb);   // Vt rows = d, cols = tokens
      o[si] = wmma_bf16(aP, bv, o[si]);
    }
    __syncthreads();
  }

  // Epilogue: O /= l, scatter per reference reshape chain
#pragma unroll
  for (int j = 0; j < 8; ++j) {
    const float invl = 1.0f / Lr[j];
    const int q = q0 + j + ph * 8;
    const int hb2 = h * NBATCH + b;
    const int b1 = hb2 / NHEAD, h1 = hb2 % NHEAD;
#pragma unroll
    for (int si = 0; si < 4; ++si) {
      const int d = si * 16 + li;
      const size_t flat = (((size_t)(h1 * SEQ + q)) * NBATCH + b1) * HDIM + d;
      concat[flat] = o[si][j] * invl;
    }
  }
}

__global__ __launch_bounds__(256) void ln_kernel(
    const float* __restrict__ x, const float* __restrict__ resid,
    const float* __restrict__ g, const float* __restrict__ bb,
    float* __restrict__ out) {
  __shared__ float sa[256], sb[256];
  const int r = blockIdx.x;
  const int t = threadIdx.x;
  const float* xr = x + (size_t)r * DIM;
  const float* qr = resid + (size_t)r * DIM;
  float a = 0.f, a2 = 0.f;
  for (int d = t; d < DIM; d += 256) { const float v = xr[d] + qr[d]; a += v; a2 += v * v; }
  sa[t] = a; sb[t] = a2; __syncthreads();
  for (int s = 128; s > 0; s >>= 1) {
    if (t < s) { sa[t] += sa[t + s]; sb[t] += sb[t + s]; }
    __syncthreads();
  }
  const float mu   = sa[0] * (1.0f / DIM);
  const float var  = sb[0] * (1.0f / DIM) - mu * mu;
  const float rstd = rsqrtf(var + 1e-5f);
  for (int d = t; d < DIM; d += 256) {
    const float v = xr[d] + qr[d];
    out[(size_t)r * DIM + d] = g[d] * (v - mu) * rstd + bb[d];
  }
}

extern "C" void kernel_launch(void* const* d_in, const int* in_sizes, int n_in,
                              void* d_out, int out_size, void* d_ws, size_t ws_size,
                              hipStream_t stream) {
  (void)in_sizes; (void)n_in; (void)out_size; (void)ws_size;
  const float* key   = (const float*)d_in[0];
  const float* query = (const float*)d_in[1];
  const float* value = (const float*)d_in[2];
  const unsigned char* mask = (const unsigned char*)d_in[3];
  const float* Wk_w = (const float*)d_in[4];
  const float* Wk_b = (const float*)d_in[5];
  const float* Wq_w = (const float*)d_in[6];
  const float* Wq_b = (const float*)d_in[7];
  const float* Wv_w = (const float*)d_in[8];
  const float* Wv_b = (const float*)d_in[9];
  const float* Wc_w = (const float*)d_in[10];
  const float* Wc_b = (const float*)d_in[11];
  const float* ln_g = (const float*)d_in[12];
  const float* ln_b = (const float*)d_in[13];
  float* out = (float*)d_out;

  const size_t XD = (size_t)NTOK * DIM;   // 3,145,728 elems
  const size_t WW = (size_t)DIM * DIM;    //   589,824 elems
  unsigned short* Xq  = (unsigned short*)d_ws;
  unsigned short* Xk  = Xq  + XD;
  unsigned short* Xv  = Xk  + XD;
  unsigned short* Wqh = Xv  + XD;
  unsigned short* Wkh = Wqh + WW;
  unsigned short* Wvh = Wkh + WW;
  unsigned short* Wch = Wvh + WW;
  unsigned short* Qp  = Wch + WW;
  unsigned short* Kp  = Qp  + XD;
  unsigned short* Vt  = Kp  + XD;
  float*          Cc  = (float*)(Vt + XD);
  unsigned short* Cb  = (unsigned short*)(Cc + XD);
  float*          P2  = (float*)(Cb + XD);

  const int NXD = (int)XD, NWW = (int)WW;
  f2bf_kernel<<<(NXD + 255) / 256, 256, 0, stream>>>(query, Xq, NXD);
  f2bf_kernel<<<(NXD + 255) / 256, 256, 0, stream>>>(key,   Xk, NXD);
  f2bf_kernel<<<(NXD + 255) / 256, 256, 0, stream>>>(value, Xv, NXD);
  f2bf_kernel<<<(NWW + 255) / 256, 256, 0, stream>>>(Wq_w, Wqh, NWW);
  f2bf_kernel<<<(NWW + 255) / 256, 256, 0, stream>>>(Wk_w, Wkh, NWW);
  f2bf_kernel<<<(NWW + 255) / 256, 256, 0, stream>>>(Wv_w, Wvh, NWW);
  f2bf_kernel<<<(NWW + 255) / 256, 256, 0, stream>>>(Wc_w, Wch, NWW);

  dim3 ggrid(NTOK / 64, DIM / 256);
  gemm_bf16_kernel<<<ggrid, 256, 0, stream>>>(Xq, Wqh, Wq_b, nullptr, Qp, 0);
  gemm_bf16_kernel<<<ggrid, 256, 0, stream>>>(Xk, Wkh, Wk_b, nullptr, Kp, 0);
  gemm_bf16_kernel<<<ggrid, 256, 0, stream>>>(Xv, Wvh, Wv_b, nullptr, Vt, 1);

  attn_kernel<<<NHEAD * NBATCH * (SEQ / 64), 128, 0, stream>>>(Qp, Kp, Vt, mask, Cc);

  f2bf_kernel<<<(NXD + 255) / 256, 256, 0, stream>>>(Cc, Cb, NXD);
  gemm_bf16_kernel<<<ggrid, 256, 0, stream>>>(Cb, Wch, Wc_b, P2, nullptr, 2);

  ln_kernel<<<NTOK, 256, 0, stream>>>(P2, query, ln_g, ln_b, out);
}
